// ChildSumTreeLSTM_24086176596338
// MI455X (gfx1250) — matrix operand order repaired and checked by reading, not compiled, not run on hardware
//
#include <hip/hip_runtime.h>
#include <hip/hip_bf16.h>

typedef __attribute__((ext_vector_type(16))) _Float16 v16h;
typedef __attribute__((ext_vector_type(8)))  _Float16 v8h;
typedef __attribute__((ext_vector_type(8)))  float    v8f;

static constexpr int Bb = 32;     // batch
static constexpr int Nn = 63;     // tree nodes
static constexpr int Hh = 1024;   // hidden = embed
static constexpr int Ll = 5;      // logits
static constexpr int NB = Nn * Bb;        // 2016 rows, node-major
static constexpr int BH = Bb * Hh;        // 32768

__device__ __forceinline__ float sigmoidf_(float x) {
    return 1.0f / (1.0f + __expf(-x));
}

// ---------------------------------------------------------------------------
// f32 -> f16 conversion (weights)
// ---------------------------------------------------------------------------
__global__ void f32_to_f16_kernel(const float* __restrict__ src,
                                  _Float16* __restrict__ dst, int n) {
    int i = blockIdx.x * blockDim.x + threadIdx.x;
    if (i < n) dst[i] = (_Float16)src[i];
}

// ---------------------------------------------------------------------------
// Embedding gather: X16[(n*B+b)*H + e] = (f16) embed[tokens[b*N+n]*H + e]
// ---------------------------------------------------------------------------
__global__ void gather_embed_kernel(const int* __restrict__ tokens,
                                    const float* __restrict__ embed,
                                    _Float16* __restrict__ X16) {
    int i = blockIdx.x * blockDim.x + threadIdx.x;
    if (i >= NB * Hh) return;
    int e  = i & (Hh - 1);
    int nb = i >> 10;            // n*B + b
    int b  = nb & (Bb - 1);
    int n  = nb >> 5;
    int tok = tokens[b * Nn + n];
    X16[i] = (_Float16)embed[(size_t)tok * Hh + e];
}

// ---------------------------------------------------------------------------
// WMMA GEMM: C[rows x 1024] = A16[rows x 1024] @ W16[1024 x 1024]^T (+ bias)
// Block = 128 threads = 4 waves along N. Block tile 32 rows x 256 cols,
// wave tile 32 x 64 (2 A-frags x 4 B-frags -> 8 v_wmma per K step).
// rows must be a multiple of 32; grid = (rows/32, 4).
// Fragment layouts per CDNA5 ISA 7.12.2 (16-bit A 16x32, B 32x16, f32 C/D).
// ---------------------------------------------------------------------------
__global__ __launch_bounds__(128) void gemm_f16_wmma_kernel(
    const _Float16* __restrict__ A,   // rows x 1024, row-major
    const _Float16* __restrict__ W,   // 1024 x 1024, row-major (out-dim major)
    const float*    __restrict__ bias,// may be null
    float*          __restrict__ C)   // rows x 1024
{
    const int lane = threadIdx.x & 31;
    const int wave = threadIdx.x >> 5;
    const int row0 = blockIdx.x * 32;
    const int col0 = blockIdx.y * 256 + wave * 64;
    const int hi   = (lane >> 4) & 1;   // 0: lanes 0-15, 1: lanes 16-31
    const int l16  = lane & 15;

    const int aoff = hi ? 16 : 0;       // A: K=0..15 vs 16..31 per lane group
    const int boff = hi ? 16 : 0;       // B: K=0..15 vs 16..31 per lane group

    v8f acc[2][4] = {};

    const _Float16* Ar0 = A + (size_t)(row0 + l16) * Hh;
    const _Float16* Ar1 = A + (size_t)(row0 + 16 + l16) * Hh;

    for (int k = 0; k < Hh; k += 32) {
        v8h a0lo = *(const v8h*)(Ar0 + k + aoff);
        v8h a0hi = *(const v8h*)(Ar0 + k + aoff + 8);
        v8h a1lo = *(const v8h*)(Ar1 + k + aoff);
        v8h a1hi = *(const v8h*)(Ar1 + k + aoff + 8);
        v16h a0 = __builtin_shufflevector(a0lo, a0hi,
                    0,1,2,3,4,5,6,7,8,9,10,11,12,13,14,15);
        v16h a1 = __builtin_shufflevector(a1lo, a1hi,
                    0,1,2,3,4,5,6,7,8,9,10,11,12,13,14,15);
#pragma unroll
        for (int f = 0; f < 4; ++f) {
            v16h bf = *(const v16h*)(W + (size_t)(col0 + f * 16 + l16) * Hh
                                       + k + boff);
            acc[0][f] = __builtin_amdgcn_wmma_f32_16x16x32_f16(
                false, a0, false, bf, (short)0, acc[0][f], false, false);
            acc[1][f] = __builtin_amdgcn_wmma_f32_16x16x32_f16(
                false, a1, false, bf, (short)0, acc[1][f], false, false);
        }
    }

    const int mb = hi ? 8 : 0;
#pragma unroll
    for (int i = 0; i < 2; ++i) {
#pragma unroll
        for (int f = 0; f < 4; ++f) {
            int col = col0 + f * 16 + l16;
            float bv = bias ? bias[col] : 0.0f;
#pragma unroll
            for (int j = 0; j < 8; ++j) {
                C[(size_t)(row0 + i * 16 + mb + j) * Hh + col] = acc[i][f][j] + bv;
            }
        }
    }
}

// ---------------------------------------------------------------------------
// Leaf gates: nodes 31..62 (d = 5). c = i*u, h = o*tanh(c).
// ---------------------------------------------------------------------------
__global__ void leaf_gates_kernel(const float* __restrict__ xi,
                                  const float* __restrict__ xf,
                                  const float* __restrict__ xu,
                                  const float* __restrict__ bih,
                                  const float* __restrict__ bfh,
                                  const float* __restrict__ buh,
                                  float* __restrict__ cbuf,
                                  float* __restrict__ hbuf,
                                  _Float16* __restrict__ h16) {
    int i = blockIdx.x * blockDim.x + threadIdx.x;   // 32 nodes * B * H
    if (i >= 32 * BH) return;
    int t = i & (Hh - 1);
    size_t idx = (size_t)(31 * Bb) * Hh + i;         // leaf rows start at node 31
    float iv = sigmoidf_(xi[idx] + bih[t]);
    float ov = sigmoidf_(xf[idx] + bfh[t]);
    float uv = tanhf(xu[idx] + buh[t]);
    float cv = iv * uv;
    float hv = ov * tanhf(cv);
    cbuf[idx] = cv;
    hbuf[idx] = hv;
    h16[idx]  = (_Float16)hv;
}

// ---------------------------------------------------------------------------
// hsum16[(j*B+b)*H + t] = (f16)(h[child0] + h[child1]) for level with m parents
// ---------------------------------------------------------------------------
__global__ void hsum_kernel(const float* __restrict__ hbuf,
                            _Float16* __restrict__ hsum16, int m, int cs) {
    int i = blockIdx.x * blockDim.x + threadIdx.x;
    if (i >= m * BH) return;
    int t  = i & (Hh - 1);
    int jb = i >> 10;
    int b  = jb & (Bb - 1);
    int j  = jb >> 5;
    size_t c0 = ((size_t)(cs + 2 * j) * Bb + b) * Hh + t;
    hsum16[i] = (_Float16)(hbuf[c0] + hbuf[c0 + BH]);
}

// ---------------------------------------------------------------------------
// Internal-node gates. gi/gu are level-local (m*B rows); yc covers the whole
// child level (2m*B rows). f-gates reuse yc per child; o-gate uses y0+y1.
// ---------------------------------------------------------------------------
__global__ void node_gates_kernel(const float* __restrict__ xi,
                                  const float* __restrict__ xf,
                                  const float* __restrict__ xu,
                                  const float* __restrict__ gi,
                                  const float* __restrict__ gu,
                                  const float* __restrict__ yc,
                                  const float* __restrict__ bih,
                                  const float* __restrict__ bfh,
                                  const float* __restrict__ buh,
                                  float* __restrict__ cbuf,
                                  float* __restrict__ hbuf,
                                  _Float16* __restrict__ h16,
                                  int m, int s, int cs) {
    int i = blockIdx.x * blockDim.x + threadIdx.x;
    if (i >= m * BH) return;
    int t  = i & (Hh - 1);
    int jb = i >> 10;
    int b  = jb & (Bb - 1);
    int j  = jb >> 5;
    size_t pr = (size_t)(s * Bb) * Hh + i;               // parent element
    size_t y0 = ((size_t)(2 * j) * Bb + b) * Hh + t;     // child-local rows in yc
    size_t c0 = ((size_t)(cs + 2 * j) * Bb + b) * Hh + t;

    float xfv = xf[pr];
    float yv0 = yc[y0];
    float yv1 = yc[y0 + BH];
    float iv = sigmoidf_(xi[pr] + gi[i] + bih[t]);
    float uv = tanhf(xu[pr] + gu[i] + buh[t]);
    float ov = sigmoidf_(xfv + yv0 + yv1 + bfh[t]);
    float f0 = sigmoidf_(xfv + yv0 + bfh[t]);
    float f1 = sigmoidf_(xfv + yv1 + bfh[t]);
    float cv = iv * uv + f0 * cbuf[c0] + f1 * cbuf[c0 + BH];
    float hv = ov * tanhf(cv);
    cbuf[pr] = cv;
    hbuf[pr] = hv;
    h16[pr]  = (_Float16)hv;
}

// ---------------------------------------------------------------------------
// logits[b,l] = h[node0,b,:] . Wout[l,:] + bout[l]
// ---------------------------------------------------------------------------
__global__ void logits_kernel(const float* __restrict__ hbuf,
                              const float* __restrict__ Wout,
                              const float* __restrict__ bout,
                              float* __restrict__ out) {
    __shared__ float red[256];
    int b = blockIdx.x / Ll;
    int l = blockIdx.x % Ll;
    const float* hv = hbuf + (size_t)b * Hh;   // node 0 occupies first B*H
    const float* w  = Wout + (size_t)l * Hh;
    float p = 0.0f;
    for (int t = threadIdx.x; t < Hh; t += 256) p += hv[t] * w[t];
    red[threadIdx.x] = p;
    __syncthreads();
    for (int sft = 128; sft > 0; sft >>= 1) {
        if (threadIdx.x < sft) red[threadIdx.x] += red[threadIdx.x + sft];
        __syncthreads();
    }
    if (threadIdx.x == 0) out[BH + b * Ll + l] = red[0] + bout[l];
}

__global__ void copy_c0_kernel(const float* __restrict__ cbuf,
                               float* __restrict__ out) {
    int i = blockIdx.x * blockDim.x + threadIdx.x;
    if (i < BH) out[i] = cbuf[i];   // node 0 is the first B*H of node-major c
}

// ---------------------------------------------------------------------------
extern "C" void kernel_launch(void* const* d_in, const int* in_sizes, int n_in,
                              void* d_out, int out_size, void* d_ws, size_t ws_size,
                              hipStream_t stream) {
    const int*   tokens = (const int*)  d_in[0];
    const float* embed  = (const float*)d_in[1];
    const float* Wix  = (const float*)d_in[2];
    const float* bix  = (const float*)d_in[3];
    const float* Wih  = (const float*)d_in[4];
    const float* bih  = (const float*)d_in[5];
    const float* Wfx  = (const float*)d_in[6];
    const float* bfx  = (const float*)d_in[7];
    const float* Wfh  = (const float*)d_in[8];
    const float* bfh  = (const float*)d_in[9];
    const float* Wux  = (const float*)d_in[10];
    const float* bux  = (const float*)d_in[11];
    const float* Wuh  = (const float*)d_in[12];
    const float* buh  = (const float*)d_in[13];
    const float* Wout = (const float*)d_in[14];
    const float* bout = (const float*)d_in[15];
    float* out = (float*)d_out;

    // ---- workspace carve-up -------------------------------------------------
    char*  base = (char*)d_ws;
    size_t off  = 0;
    auto carve = [&](size_t bytes) -> void* {
        void* p = base + off;
        off = (off + bytes + 255) & ~(size_t)255;
        return p;
    };
    const size_t WElems = (size_t)Hh * Hh;
    _Float16* Wix16 = (_Float16*)carve(WElems * 2);
    _Float16* Wih16 = (_Float16*)carve(WElems * 2);
    _Float16* Wfx16 = (_Float16*)carve(WElems * 2);
    _Float16* Wfh16 = (_Float16*)carve(WElems * 2);
    _Float16* Wux16 = (_Float16*)carve(WElems * 2);
    _Float16* Wuh16 = (_Float16*)carve(WElems * 2);
    _Float16* X16   = (_Float16*)carve((size_t)NB * Hh * 2);
    float* xi = (float*)carve((size_t)NB * Hh * 4);
    float* xf = (float*)carve((size_t)NB * Hh * 4);
    float* xu = (float*)carve((size_t)NB * Hh * 4);
    float* hbuf = (float*)carve((size_t)NB * Hh * 4);
    float* cbuf = (float*)carve((size_t)NB * Hh * 4);
    _Float16* h16    = (_Float16*)carve((size_t)NB * Hh * 2);
    _Float16* hsum16 = (_Float16*)carve((size_t)16 * BH * 2);   // max m = 16
    float* gi = (float*)carve((size_t)16 * BH * 4);
    float* gu = (float*)carve((size_t)16 * BH * 4);
    float* yc = (float*)carve((size_t)32 * BH * 4);             // max 2m = 32

    const int T = 256;
    // ---- weight conversion to f16 ------------------------------------------
    int wgrid = (int)((WElems + T - 1) / T);
    f32_to_f16_kernel<<<wgrid, T, 0, stream>>>(Wix, Wix16, (int)WElems);
    f32_to_f16_kernel<<<wgrid, T, 0, stream>>>(Wih, Wih16, (int)WElems);
    f32_to_f16_kernel<<<wgrid, T, 0, stream>>>(Wfx, Wfx16, (int)WElems);
    f32_to_f16_kernel<<<wgrid, T, 0, stream>>>(Wfh, Wfh16, (int)WElems);
    f32_to_f16_kernel<<<wgrid, T, 0, stream>>>(Wux, Wux16, (int)WElems);
    f32_to_f16_kernel<<<wgrid, T, 0, stream>>>(Wuh, Wuh16, (int)WElems);

    // ---- embedding gather ---------------------------------------------------
    gather_embed_kernel<<<(NB * Hh + T - 1) / T, T, 0, stream>>>(tokens, embed, X16);

    // ---- input projections (rows = 2016) -----------------------------------
    dim3 gproj(NB / 32, 4);
    gemm_f16_wmma_kernel<<<gproj, 128, 0, stream>>>(X16, Wix16, bix, xi);
    gemm_f16_wmma_kernel<<<gproj, 128, 0, stream>>>(X16, Wfx16, bfx, xf);
    gemm_f16_wmma_kernel<<<gproj, 128, 0, stream>>>(X16, Wux16, bux, xu);

    // ---- leaf level (d = 5) -------------------------------------------------
    leaf_gates_kernel<<<(32 * BH + T - 1) / T, T, 0, stream>>>(
        xi, xf, xu, bih, bfh, buh, cbuf, hbuf, h16);

    // ---- internal levels d = 4 .. 0 ----------------------------------------
    for (int d = 4; d >= 0; --d) {
        int m  = 1 << d;
        int s  = m - 1;        // first node of this level
        int cs = 2 * m - 1;    // first node of child level
        int rows_p = m * Bb;   // parent rows
        int rows_c = 2 * m * Bb;

        hsum_kernel<<<(m * BH + T - 1) / T, T, 0, stream>>>(hbuf, hsum16, m, cs);

        dim3 gp(rows_p / 32, 4), gc(rows_c / 32, 4);
        gemm_f16_wmma_kernel<<<gp, 128, 0, stream>>>(hsum16, Wih16, nullptr, gi);
        gemm_f16_wmma_kernel<<<gp, 128, 0, stream>>>(hsum16, Wuh16, nullptr, gu);
        gemm_f16_wmma_kernel<<<gc, 128, 0, stream>>>(
            h16 + (size_t)cs * Bb * Hh, Wfh16, nullptr, yc);

        node_gates_kernel<<<(m * BH + T - 1) / T, T, 0, stream>>>(
            xi, xf, xu, gi, gu, yc, bih, bfh, buh, cbuf, hbuf, h16, m, s, cs);
    }

    // ---- outputs ------------------------------------------------------------
    copy_c0_kernel<<<(BH + T - 1) / T, T, 0, stream>>>(cbuf, out);
    logits_kernel<<<Bb * Ll, 256, 0, stream>>>(hbuf, Wout, bout, out);
}
